// EngramModule_35553739276786
// MI455X (gfx1250) — matrix operand order repaired and checked by reading, not compiled, not run on hardware
//
#include <hip/hip_runtime.h>
#include <hip/hip_bf16.h>

// ---------------------------------------------------------------------------
// Engram memory-attention block for gfx1250 (MI455X), bf16 WMMA GEMM path,
// double-buffered LDS with async global->LDS staging for the weight tile.
// ---------------------------------------------------------------------------

#define BDIM 1024
#define NHEADS 16
#define DHEAD 64
#define TOPK 8

#define BM 128
#define BN 128
#define BK 32
#define LDSP (BK + 8)   // padded LDS row (bf16 elems): 80B stride, conflict-free

#if defined(__AMDGCN__) && \
    __has_builtin(__builtin_amdgcn_global_load_async_to_lds_b128) && \
    __has_builtin(__builtin_amdgcn_s_wait_asynccnt)
#define HAVE_ASYNC_LDS 1
#else
#define HAVE_ASYNC_LDS 0
#endif

typedef __attribute__((ext_vector_type(16))) __bf16 v16bf;
typedef __attribute__((ext_vector_type(4)))  __bf16 v4bf;
typedef __attribute__((ext_vector_type(8)))  float  v8f;
typedef __attribute__((ext_vector_type(4)))  int    v4i;
#if HAVE_ASYNC_LDS
typedef __attribute__((address_space(1))) v4i v4i_g;   // global b128 pointee
typedef __attribute__((address_space(3))) v4i v4i_l;   // LDS b128 pointee
#endif

union Frag32 { v16bf v; uint4 q[2]; };

// ---------------------------------------------------------------------------
// Weight transpose + f32->bf16 convert:  W[R][C] (f32) -> Wt[C][R] (bf16)
// ---------------------------------------------------------------------------
__global__ __launch_bounds__(256)
void transpose_to_bf16(const float* __restrict__ W, __bf16* __restrict__ Wt,
                       int R, int C) {
    __shared__ float tile[32][33];
    int c0 = blockIdx.x * 32, r0 = blockIdx.y * 32;
    int tx = threadIdx.x, ty = threadIdx.y;          // 32 x 8
    #pragma unroll
    for (int i = 0; i < 32; i += 8)
        tile[ty + i][tx] = W[(size_t)(r0 + ty + i) * C + c0 + tx];
    __syncthreads();
    #pragma unroll
    for (int i = 0; i < 32; i += 8)
        Wt[(size_t)(c0 + ty + i) * R + r0 + tx] = (__bf16)tile[tx][ty + i];
}

// ---------------------------------------------------------------------------
// Tiled GEMM: C[M][N] = A[M][K] (f32 -> bf16 on stage) x Bt[N][K] (bf16)
// Dual-A: columns k < kSplit read from A0 (lda=kSplit), else A1 (lda=K-kSplit)
// 256 threads = 8 waves; wave -> 32x64 subtile -> 8x v_wmma_f32_16x16x32_bf16
// per BK step. Double-buffered LDS, one barrier per K-step; B tile staged via
// GLOBAL_LOAD_ASYNC_TO_LDS_B128 when available. Dims divisible by tiles.
// ---------------------------------------------------------------------------
template<bool OUT_BF16>
__global__ __launch_bounds__(256)
void wmma_gemm_kernel(const float* __restrict__ A0, const float* __restrict__ A1,
                      int kSplit,
                      const __bf16* __restrict__ Bt,
                      const float* __restrict__ bias,
                      float* __restrict__ Cf, __bf16* __restrict__ Cb,
                      int M, int N, int K) {
    __shared__ alignas(16) __bf16 As[2][BM][LDSP];
    __shared__ alignas(16) __bf16 Bs[2][BN][LDSP];

    const int tid  = threadIdx.x;
    const int wave = tid >> 5;
    const int lane = tid & 31;
    const int half = lane >> 4;
    const int r16  = lane & 15;
    const int wm   = wave & 3;    // 4 row-groups of 32
    const int wn   = wave >> 2;   // 2 col-groups of 64

    const int rowBase = blockIdx.y * BM;
    const int colBase = blockIdx.x * BN;

    // ---- staging helpers -------------------------------------------------
    auto loadA = [&](int k0x, float4* pa) {
        const float* Asrc = (k0x < kSplit) ? A0 : A1;
        const int    aK   = (k0x < kSplit) ? kSplit : (K - kSplit);
        const int    akk  = (k0x < kSplit) ? k0x : (k0x - kSplit);
        #pragma unroll
        for (int i = 0; i < 4; ++i) {
            int ci = tid + i * 256;
            int r  = ci >> 3;
            int c4 = (ci & 7) << 2;
            pa[i] = *(const float4*)(Asrc + (size_t)(rowBase + r) * aK + akk + c4);
        }
    };
    auto storeA = [&](int buf, const float4* pa) {
        #pragma unroll
        for (int i = 0; i < 4; ++i) {
            int ci = tid + i * 256;
            int r  = ci >> 3;
            int c4 = (ci & 7) << 2;
            // native packed converts: 2x v_cvt_pk_bf16_f32 + ds_store_b64
            v4bf p = { (__bf16)pa[i].x, (__bf16)pa[i].y,
                       (__bf16)pa[i].z, (__bf16)pa[i].w };
            *(v4bf*)(&As[buf][r][c4]) = p;
        }
    };
#if HAVE_ASYNC_LDS
    auto stageB = [&](int k0x, int buf) {    // global bf16 -> LDS, no VGPR data
        #pragma unroll
        for (int i = 0; i < 2; ++i) {
            int ci = tid + i * 256;
            int r  = ci >> 2;
            int c8 = (ci & 3) << 3;
            const __bf16* g = Bt + (size_t)(colBase + r) * K + k0x + c8;
            __builtin_amdgcn_global_load_async_to_lds_b128(
                (v4i_g*)(void*)g,
                (v4i_l*)(void*)&Bs[buf][r][c8],
                0, 0);
        }
    };
#else
    auto loadB = [&](int k0x, uint4* pb) {
        #pragma unroll
        for (int i = 0; i < 2; ++i) {
            int ci = tid + i * 256;
            int r  = ci >> 2;
            int c8 = (ci & 3) << 3;
            pb[i] = *(const uint4*)(Bt + (size_t)(colBase + r) * K + k0x + c8);
        }
    };
    auto storeB = [&](int buf, const uint4* pb) {
        #pragma unroll
        for (int i = 0; i < 2; ++i) {
            int ci = tid + i * 256;
            int r  = ci >> 2;
            int c8 = (ci & 3) << 3;
            *(uint4*)(&Bs[buf][r][c8]) = pb[i];
        }
    };
#endif

    v8f acc[2][4];
    #pragma unroll
    for (int m = 0; m < 2; ++m)
        #pragma unroll
        for (int n = 0; n < 4; ++n)
            #pragma unroll
            for (int d = 0; d < 8; ++d) acc[m][n][d] = 0.f;

    const int nk = K / BK;

    // ---- prologue: stage tile 0 into buffer 0 ----
    {
        float4 pa[4];
        loadA(0, pa);
#if HAVE_ASYNC_LDS
        stageB(0, 0);
#else
        uint4 pb[2];
        loadB(0, pb);
        storeB(0, pb);
#endif
        storeA(0, pa);
#if HAVE_ASYNC_LDS
        __builtin_amdgcn_s_wait_asynccnt(0);
#endif
        __syncthreads();
    }

    for (int t = 0; t < nk; ++t) {
        const int cur = t & 1, nxt = cur ^ 1;
        const bool more = (t + 1) < nk;

        // ---- issue next-tile global loads early (overlap with WMMA) ----
        float4 pa[4];
#if !HAVE_ASYNC_LDS
        uint4 pb[2];
#endif
        if (more) {
            const int k0n = (t + 1) * BK;
#if HAVE_ASYNC_LDS
            stageB(k0n, nxt);           // async: writes LDS directly
#else
            loadB(k0n, pb);
#endif
            loadA(k0n, pa);
        }

        // ---- fragments per ISA wave32 layouts ----
        Frag32 af[2], bf[4];
        #pragma unroll
        for (int m = 0; m < 2; ++m) {
            int row = wm * 32 + m * 16 + r16;
            int ka  = half * 8;          // lanes0-15: K0-7/K16-23; 16-31: +8
            af[m].q[0] = *(const uint4*)(&As[cur][row][ka]);
            af[m].q[1] = *(const uint4*)(&As[cur][row][ka + 16]);
        }
        #pragma unroll
        for (int n = 0; n < 4; ++n) {
            int col = wn * 64 + n * 16 + r16;
            int kb  = half * 8;          // lanes0-15: K0-7/K16-23; 16-31: +8
            bf[n].q[0] = *(const uint4*)(&Bs[cur][col][kb]);
            bf[n].q[1] = *(const uint4*)(&Bs[cur][col][kb + 16]);
        }
        #pragma unroll
        for (int m = 0; m < 2; ++m)
            #pragma unroll
            for (int n = 0; n < 4; ++n)
                acc[m][n] = __builtin_amdgcn_wmma_f32_16x16x32_bf16(
                    false, af[m].v, false, bf[n].v,
                    (short)0, acc[m][n], false, false);

        // ---- commit next tile into the other buffer ----
        if (more) {
            storeA(nxt, pa);
#if !HAVE_ASYNC_LDS
            storeB(nxt, pb);
#endif
        }
#if HAVE_ASYNC_LDS
        if (more) __builtin_amdgcn_s_wait_asynccnt(0);
#endif
        __syncthreads();
    }

    // ---- epilogue: C VGPR d -> row (half?8:0)+d, col r16 ----
    #pragma unroll
    for (int m = 0; m < 2; ++m) {
        #pragma unroll
        for (int n = 0; n < 4; ++n) {
            int colg = colBase + wn * 64 + n * 16 + r16;
            float bv = bias ? bias[colg] : 0.f;
            #pragma unroll
            for (int d = 0; d < 8; ++d) {
                int rowg = rowBase + wm * 32 + m * 16 + half * 8 + d;
                float v  = acc[m][n][d] + bv;
                size_t idx = (size_t)rowg * N + colg;
                if (OUT_BF16) Cb[idx] = (__bf16)v;
                else          Cf[idx] = v;
            }
        }
    }
}

// ---------------------------------------------------------------------------
// Attention over TOPK=8 memory slots. One wave32 per (b, head); lane owns
// dims {lane, lane+32} of DHEAD=64. Scores all-reduced with shfl_xor.
// ---------------------------------------------------------------------------
__global__ __launch_bounds__(256)
void attn_kernel(const float* __restrict__ Q, const __bf16* __restrict__ Kp,
                 const __bf16* __restrict__ Vp, float* __restrict__ O) {
    int warp = threadIdx.x >> 5, lane = threadIdx.x & 31;
    int gid = blockIdx.x * 8 + warp;      // b*NHEADS + h
    int b = gid >> 4, h = gid & 15;
    const float scale = 0.125f;           // 64^-0.5

    size_t qoff = (size_t)b * BDIM + h * DHEAD;
    float q0 = Q[qoff + lane], q1 = Q[qoff + lane + 32];

    float v0[TOPK], v1[TOPK], sc[TOPK];
    #pragma unroll
    for (int k = 0; k < TOPK; ++k) {
        size_t kv = ((size_t)(b * TOPK + k)) * BDIM + h * DHEAD;
        float k0 = (float)Kp[kv + lane], k1 = (float)Kp[kv + lane + 32];
        v0[k] = (float)Vp[kv + lane];
        v1[k] = (float)Vp[kv + lane + 32];
        float s = q0 * k0 + q1 * k1;
        #pragma unroll
        for (int off = 16; off; off >>= 1) s += __shfl_xor(s, off, 32);
        sc[k] = s * scale;
    }
    float mx = sc[0];
    #pragma unroll
    for (int k = 1; k < TOPK; ++k) mx = fmaxf(mx, sc[k]);
    float sum = 0.f;
    #pragma unroll
    for (int k = 0; k < TOPK; ++k) { sc[k] = __expf(sc[k] - mx); sum += sc[k]; }
    float inv = 1.f / sum;
    float o0 = 0.f, o1 = 0.f;
    #pragma unroll
    for (int k = 0; k < TOPK; ++k) {
        float a = sc[k] * inv;
        o0 += a * v0[k];
        o1 += a * v1[k];
    }
    O[qoff + lane] = o0;
    O[qoff + lane + 32] = o1;
}

// ---------------------------------------------------------------------------
// Fused gate(sigmoid) + residual + LayerNorm. One block per row (1024 elems).
// ---------------------------------------------------------------------------
__global__ __launch_bounds__(256)
void fuse_ln_kernel(const float* __restrict__ H, const float* __restrict__ MO,
                    const float* __restrict__ GL, const float* __restrict__ lng,
                    const float* __restrict__ lnb, float* __restrict__ out) {
    int t = threadIdx.x, lane = t & 31, warp = t >> 5;
    size_t base = (size_t)blockIdx.x * BDIM;

    float a[4], s1 = 0.f, s2 = 0.f;
    #pragma unroll
    for (int i = 0; i < 4; ++i) {
        int idx = t + i * 256;
        float h  = H[base + idx];
        float mo = MO[base + idx];
        float g  = 1.f / (1.f + __expf(-GL[base + idx]));
        float v  = h + g * mo;
        a[i] = v; s1 += v; s2 += v * v;
    }
    #pragma unroll
    for (int off = 16; off; off >>= 1) {
        s1 += __shfl_xor(s1, off, 32);
        s2 += __shfl_xor(s2, off, 32);
    }
    __shared__ float r1[8], r2[8];
    if (lane == 0) { r1[warp] = s1; r2[warp] = s2; }
    __syncthreads();
    float t1 = 0.f, t2 = 0.f;
    #pragma unroll
    for (int w = 0; w < 8; ++w) { t1 += r1[w]; t2 += r2[w]; }
    float mean = t1 * (1.f / 1024.f);
    float var  = t2 * (1.f / 1024.f) - mean * mean;
    float rstd = rsqrtf(var + 1e-5f);
    #pragma unroll
    for (int i = 0; i < 4; ++i) {
        int idx = t + i * 256;
        out[base + idx] = (a[i] - mean) * rstd * lng[idx] + lnb[idx];
    }
}

// ---------------------------------------------------------------------------
extern "C" void kernel_launch(void* const* d_in, const int* in_sizes, int n_in,
                              void* d_out, int out_size, void* d_ws, size_t ws_size,
                              hipStream_t stream) {
    const float* H   = (const float*)d_in[0];   // (8192, 1024)
    const float* MK  = (const float*)d_in[1];   // (8192, 8, 1024) -> (65536,1024)
    const float* MV  = (const float*)d_in[2];   // (8192, 8, 1024)
    const float* Wq  = (const float*)d_in[3];
    const float* Wk  = (const float*)d_in[4];
    const float* Wv  = (const float*)d_in[5];
    const float* Wo  = (const float*)d_in[6];
    const float* Wg  = (const float*)d_in[7];   // (2048, 1024)
    const float* bg  = (const float*)d_in[8];
    const float* lng = (const float*)d_in[9];
    const float* lnb = (const float*)d_in[10];
    float* out = (float*)d_out;

    char* ws = (char*)d_ws;
    const size_t MiB = 1024u * 1024u;
    __bf16* wqT  = (__bf16*)(ws + 0 * MiB);     //  2 MiB
    __bf16* wkT  = (__bf16*)(ws + 2 * MiB);     //  2 MiB
    __bf16* wvT  = (__bf16*)(ws + 4 * MiB);     //  2 MiB
    __bf16* woT  = (__bf16*)(ws + 6 * MiB);     //  2 MiB
    __bf16* wgT  = (__bf16*)(ws + 8 * MiB);     //  4 MiB  [1024][2048]
    float*  Qb   = (float*)(ws + 12 * MiB);     // 32 MiB
    __bf16* Kp   = (__bf16*)(ws + 44 * MiB);    // 128 MiB
    __bf16* Vp   = (__bf16*)(ws + 172 * MiB);   // 128 MiB
    float*  attn = (float*)(ws + 300 * MiB);    // 32 MiB
    float*  mo   = (float*)(ws + 12 * MiB);     // alias Q (dead after attn)
    float*  glin = (float*)(ws + 44 * MiB);     // alias Kp (dead after attn)

    dim3 tb(32, 8);
    transpose_to_bf16<<<dim3(32, 32), tb, 0, stream>>>(Wq, wqT, 1024, 1024);
    transpose_to_bf16<<<dim3(32, 32), tb, 0, stream>>>(Wk, wkT, 1024, 1024);
    transpose_to_bf16<<<dim3(32, 32), tb, 0, stream>>>(Wv, wvT, 1024, 1024);
    transpose_to_bf16<<<dim3(32, 32), tb, 0, stream>>>(Wo, woT, 1024, 1024);
    transpose_to_bf16<<<dim3(32, 64), tb, 0, stream>>>(Wg, wgT, 2048, 1024);

    // Q = H @ Wq                                  (8192 x 1024 x 1024)
    wmma_gemm_kernel<false><<<dim3(8, 64), 256, 0, stream>>>(
        H, H, 1024, wqT, nullptr, Qb, nullptr, 8192, 1024, 1024);
    // Kproj = MK @ Wk  -> bf16                    (65536 x 1024 x 1024)
    wmma_gemm_kernel<true><<<dim3(8, 512), 256, 0, stream>>>(
        MK, MK, 1024, wkT, nullptr, nullptr, Kp, 65536, 1024, 1024);
    // Vproj = MV @ Wv  -> bf16
    wmma_gemm_kernel<true><<<dim3(8, 512), 256, 0, stream>>>(
        MV, MV, 1024, wvT, nullptr, nullptr, Vp, 65536, 1024, 1024);
    // softmax attention over 8 slots
    attn_kernel<<<16384, 256, 0, stream>>>(Qb, Kp, Vp, attn);
    // memory_out = attn @ Wo
    wmma_gemm_kernel<false><<<dim3(8, 64), 256, 0, stream>>>(
        attn, attn, 1024, woT, nullptr, mo, nullptr, 8192, 1024, 1024);
    // gate linear = [H | mo] @ Wg + bg            (K = 2048, split at 1024)
    wmma_gemm_kernel<false><<<dim3(8, 64), 256, 0, stream>>>(
        H, mo, 1024, wgT, bg, glin, nullptr, 8192, 1024, 2048);
    // out = LN(H + sigmoid(glin) * mo)
    fuse_ln_kernel<<<8192, 256, 0, stream>>>(H, mo, glin, lng, lnb, out);

    (void)in_sizes; (void)n_in; (void)out_size; (void)ws_size;
}
